// NeighbourhoodSampling_29703993819759
// MI455X (gfx1250) — compile-verified, matrix operand order checked by reading, not run on hardware
//
#include <hip/hip_runtime.h>

typedef __attribute__((ext_vector_type(16))) _Float16 v16h;
typedef __attribute__((ext_vector_type(8)))  float    v8f;

// ---------------------------------------------------------------------------
// Workspace layout:
//   [0,        55296)  : folded weights, f16, [2 halves][27 k][32 lanes][16] (A-operand layout)
//   [57344,    57472)  : folded bias, 32 f32
//   [65536, +256MB)    : volume transposed to [D][H][W][C] f32 (channel-innermost)
// ---------------------------------------------------------------------------
#define WBUF_OFF  0
#define BIAS_OFF  57344
#define VOLT_OFF  65536

// A/B operand lane mapping for 16-bit 16x16x32 WMMA (CDNA5 ISA):
//   lo lanes (0-15) hold K in {0..7, 16..23}; hi lanes (16-31) hold K in {8..15, 24..31}
__device__ __forceinline__ int k_halfsel(int c) { return (c >> 3) & 1; }
__device__ __forceinline__ int k_slot(int c)    { return (c & 7) | (((c >> 4) & 1) << 3); }

// ---------------------------------------------------------------------------
// Fold the whole linear network into B_k (27 x 32x32) + bias, written in
// WMMA A-operand lane order as f16.  One block, 1024 threads: thread=(o,c).
// ---------------------------------------------------------------------------
__global__ void prep_weights(const float* __restrict__ w_d1, const float* __restrict__ b_d1,
                             const float* __restrict__ w_d2, const float* __restrict__ b_d2,
                             const float* __restrict__ w_c1, const float* __restrict__ b_c1,
                             const float* __restrict__ w_c2, const float* __restrict__ b_c2,
                             const float* __restrict__ conv_w, const float* __restrict__ conv_b,
                             _Float16* __restrict__ wbuf, float* __restrict__ biasbuf)
{
    __shared__ float sWd[32][32];   // w_d2 @ w_d1
    __shared__ float sWc[32][32];   // w_c2 @ w_c1
    __shared__ float sbd[32];       // w_d2 @ b_d1 + b_d2
    const int t = threadIdx.x;
    const int o = t >> 5, c = t & 31;

    float wd = 0.f, wc = 0.f;
    for (int j = 0; j < 32; ++j) {
        wd += w_d2[o * 32 + j] * w_d1[j * 32 + c];
        wc += w_c2[o * 32 + j] * w_c1[j * 32 + c];
    }
    sWd[o][c] = wd;
    sWc[o][c] = wc;
    if (c == 0) {
        float s = 0.f;
        for (int j = 0; j < 32; ++j) s += w_d2[o * 32 + j] * b_d1[j];
        sbd[o] = s + b_d2[o];
    }
    __syncthreads();

    const int m    = o >> 4;
    const int lane = (o & 15) | (k_halfsel(c) << 4);
    const int slot = k_slot(c);

    float ssum = 0.f, a13 = 0.f;
    for (int k = 0; k < 27; ++k) {
        float a = 0.f;
        for (int j = 0; j < 32; ++j)
            a += conv_w[(o * 32 + j) * 27 + k] * sWd[j][c];   // A_k = conv_w[:, :, k] @ Wd
        ssum += a;
        if (k == 13) a13 = a;
        else wbuf[((m * 27 + k) * 32 + lane) * 16 + slot] = (_Float16)a;
    }
    // B_13 = A_13 - sum_k A_k + Wc   (diff subtracts center; center branch adds Wc)
    wbuf[((m * 27 + 13) * 32 + lane) * 16 + slot] = (_Float16)(a13 - ssum + sWc[o][c]);

    if (c == 0) {
        float bb = conv_b[o];
        for (int j = 0; j < 32; ++j) {
            float cj = 0.f;
            for (int k = 0; k < 27; ++k) cj += conv_w[(o * 32 + j) * 27 + k];
            bb += cj * sbd[j];
        }
        float bc = b_c2[o];
        for (int j = 0; j < 32; ++j) bc += w_c2[o * 32 + j] * b_c1[j];
        biasbuf[o] = bb + bc;
    }
}

// ---------------------------------------------------------------------------
// [C,D,H,W] -> [D,H,W,C]: makes every trilinear corner a single coalesced
// 128B wave32 load (lane = channel).  Writes fully coalesced.
// ---------------------------------------------------------------------------
__global__ void transpose_vol(const float* __restrict__ vol, float* __restrict__ volT)
{
    long long t   = (long long)blockIdx.x * 256 + threadIdx.x;
    int       c   = (int)(t & 31);
    long long vox = t >> 5;                       // 0 .. 128^3-1
    volT[(vox << 5) + c] = vol[((long long)c << 21) + vox];
}

// ---------------------------------------------------------------------------
// Fused sampler + WMMA GEMM.  128 threads = 4 waves; each wave owns 16
// vertices.  Per neighbor k: sample 32ch x 16verts into an LDS slice laid
// out as the WMMA B operand, then two v_wmma_f32_16x16x32_f16 accumulate
// the 32 output channels over the 27-neighbor reduction.
// ---------------------------------------------------------------------------
__global__ void __launch_bounds__(128) sample_fuse(
    const float* __restrict__ volT, const float* __restrict__ verts,
    const _Float16* __restrict__ wbuf, const float* __restrict__ biasbuf,
    float* __restrict__ out, int nverts)
{
    __shared__ __align__(32) _Float16 fbuf[4][2][32][16];  // [wave][k parity][lane][slot]
    __shared__ float vbuf[4][16][3];

    const int lane = threadIdx.x & 31;
    const int wave = threadIdx.x >> 5;
    const int tile = blockIdx.x * 4 + wave;
    const int ntiles = (nverts + 15) >> 4;
    const bool active = tile < ntiles;
    const int base = active ? tile * 16 : 0;

    // stage the 16 vertex coordinates for this wave
    if (lane < 16) {
        int n = base + lane;
        if (n >= nverts) n = nverts - 1;
        vbuf[wave][lane][0] = verts[n * 3 + 0];
        vbuf[wave][lane][1] = verts[n * 3 + 1];
        vbuf[wave][lane][2] = verts[n * 3 + 2];
    }
    asm volatile("s_wait_dscnt 0" ::: "memory");

    const int c    = lane;                  // lane = channel during sampling
    const int lsel = k_halfsel(c) << 4;
    const int slot = k_slot(c);

    v8f acc0 = {};  // output channels 0..15
    v8f acc1 = {};  // output channels 16..31

    for (int k = 0; k < 27; ++k) {
        // shift = (a[k/9], a[(k/3)%3], a[k%3]) * 2^(STEPS+1-STEP)/128 = *0.0625
        const float sx = ((float)(k / 9)       - 1.f) * 0.0625f;
        const float sy = ((float)((k / 3) % 3) - 1.f) * 0.0625f;
        const float sz = ((float)(k % 3)       - 1.f) * 0.0625f;
        _Float16* slice = &fbuf[wave][k & 1][0][0];

        for (int n = 0; n < 16; ++n) {
            float px = fminf(fmaxf((vbuf[wave][n][0] + sx + 1.f) * 63.5f, 0.f), 127.f);
            float py = fminf(fmaxf((vbuf[wave][n][1] + sy + 1.f) * 63.5f, 0.f), 127.f);
            float pz = fminf(fmaxf((vbuf[wave][n][2] + sz + 1.f) * 63.5f, 0.f), 127.f);
            float fx = floorf(px), fy = floorf(py), fz = floorf(pz);
            int x0 = (int)fx, y0 = (int)fy, z0 = (int)fz;
            float wx = px - fx, wy = py - fy, wz = pz - fz;
            long dx = (long)(min(x0 + 1, 127) - x0) << 5;    // +1 voxel in x = 32 floats
            long dy = (long)(min(y0 + 1, 127) - y0) << 12;   // +1 in y = 128*32
            long dz = (long)(min(z0 + 1, 127) - z0) << 19;   // +1 in z = 128*128*32
            const float* p = volT + ((((long)z0 * 128 + y0) * 128 + x0) << 5) + c;
            float v000 = p[0],       v001 = p[dx];
            float v010 = p[dy],      v011 = p[dy + dx];
            float v100 = p[dz],      v101 = p[dz + dx];
            float v110 = p[dz + dy], v111 = p[dz + dy + dx];
            float c00 = v000 + (v001 - v000) * wx;
            float c01 = v010 + (v011 - v010) * wx;
            float c10 = v100 + (v101 - v100) * wx;
            float c11 = v110 + (v111 - v110) * wx;
            float c0  = c00 + (c01 - c00) * wy;
            float c1  = c10 + (c11 - c10) * wy;
            float val = c0  + (c1  - c0 ) * wz;
            // scatter into WMMA B-operand lane layout: (K=c, N=n)
            slice[((n | lsel)) * 16 + slot] = (_Float16)val;
        }

        asm volatile("s_wait_dscnt 0" ::: "memory");   // per-wave LDS RAW fence
        v16h bmat = *(const v16h*)(slice + lane * 16);
        v16h a0 = *(const v16h*)(wbuf + ((0 * 27 + k) * 32 + lane) * 16);
        v16h a1 = *(const v16h*)(wbuf + ((1 * 27 + k) * 32 + lane) * 16);
        acc0 = __builtin_amdgcn_wmma_f32_16x16x32_f16(false, a0, false, bmat,
                                                      (short)0, acc0, false, false);
        acc1 = __builtin_amdgcn_wmma_f32_16x16x32_f16(false, a1, false, bmat,
                                                      (short)0, acc1, false, false);
    }

    if (active) {
        // D layout: VGPR r -> M = r + (lane>=16 ? 8 : 0), N = lane & 15
        int n  = base + (lane & 15);
        int ob = (lane >> 4) << 3;
        float* op0 = out + n * 32 + ob;
        float* op1 = out + n * 32 + 16 + ob;
        const float* bp0 = biasbuf + ob;
        const float* bp1 = biasbuf + 16 + ob;
#pragma unroll
        for (int r = 0; r < 8; ++r) op0[r] = acc0[r] + bp0[r];
#pragma unroll
        for (int r = 0; r < 8; ++r) op1[r] = acc1[r] + bp1[r];
    }
}

extern "C" void kernel_launch(void* const* d_in, const int* in_sizes, int n_in,
                              void* d_out, int out_size, void* d_ws, size_t ws_size,
                              hipStream_t stream)
{
    const float* vol    = (const float*)d_in[0];
    const float* verts  = (const float*)d_in[1];
    const float* w_d1   = (const float*)d_in[2];
    const float* b_d1   = (const float*)d_in[3];
    const float* w_d2   = (const float*)d_in[4];
    const float* b_d2   = (const float*)d_in[5];
    const float* w_c1   = (const float*)d_in[6];
    const float* b_c1   = (const float*)d_in[7];
    const float* w_c2   = (const float*)d_in[8];
    const float* b_c2   = (const float*)d_in[9];
    const float* conv_w = (const float*)d_in[10];
    const float* conv_b = (const float*)d_in[11];

    _Float16* wbuf    = (_Float16*)((char*)d_ws + WBUF_OFF);
    float*    biasbuf = (float*)((char*)d_ws + BIAS_OFF);
    float*    volT    = (float*)((char*)d_ws + VOLT_OFF);

    const int nverts = in_sizes[1] / 3;

    prep_weights<<<1, 1024, 0, stream>>>(w_d1, b_d1, w_d2, b_d2,
                                         w_c1, b_c1, w_c2, b_c2,
                                         conv_w, conv_b, wbuf, biasbuf);

    // 32 channels * 128^3 voxels / 256 threads
    transpose_vol<<<262144, 256, 0, stream>>>(vol, volT);

    const int ntiles  = (nverts + 15) / 16;
    const int nblocks = (ntiles + 3) / 4;
    sample_fuse<<<nblocks, 128, 0, stream>>>(volT, verts, wbuf, biasbuf,
                                             (float*)d_out, nverts);
}
